// CustomLlamaAttention_28698971472494
// MI455X (gfx1250) — matrix operand order repaired and checked by reading, not compiled, not run on hardware
//
#include <hip/hip_runtime.h>
#include <hip/hip_bf16.h>
#include <math.h>

// ---------------------------------------------------------------------------
// Types for CDNA5 WMMA
// ---------------------------------------------------------------------------
typedef __attribute__((ext_vector_type(16))) __bf16        v16bf;
typedef __attribute__((ext_vector_type(8)))  float         v8f;
typedef __attribute__((ext_vector_type(4)))  unsigned int  v4u;
typedef __attribute__((ext_vector_type(8)))  unsigned int  v8u;

__device__ __forceinline__ v8f wmma_bf16(v16bf a, v16bf b, v8f c) {
  // D = A(16x32 bf16) * B(32x16 bf16) + C(16x16 f32)
  return __builtin_amdgcn_wmma_f32_16x16x32_bf16(
      false, a, false, b, (short)0, c, false, false);
}

// pack two floats into one dword of bf16 pair (lowers to v_cvt_pk_bf16_f32)
__device__ __forceinline__ unsigned int pk2(float x, float y) {
  unsigned short a = __builtin_bit_cast(unsigned short, (__bf16)x);
  unsigned short b = __builtin_bit_cast(unsigned short, (__bf16)y);
  return (unsigned int)a | ((unsigned int)b << 16);
}

// Async global->LDS copy of 16 bytes per lane (CDNA5 GLOBAL_LOAD_ASYNC_TO_LDS,
// tracked by ASYNCcnt).  ldsOff = LDS byte address (low 32 bits of generic
// shared pointer: LDS aperture keeps the offset in addr[31:0]).
__device__ __forceinline__ void async_cp16(unsigned ldsOff, const void* gptr) {
  unsigned long long ga = (unsigned long long)(uintptr_t)gptr;
  asm volatile("global_load_async_to_lds_b128 %0, %1, off"
               :: "v"(ldsOff), "v"(ga) : "memory");
}
__device__ __forceinline__ void async_cp64(unsigned ldsOff, const void* gptr) {
  unsigned long long ga = (unsigned long long)(uintptr_t)gptr;
  asm volatile("global_load_async_to_lds_b128 %0, %1, off"           :: "v"(ldsOff), "v"(ga) : "memory");
  asm volatile("global_load_async_to_lds_b128 %0, %1, off offset:16" :: "v"(ldsOff), "v"(ga) : "memory");
  asm volatile("global_load_async_to_lds_b128 %0, %1, off offset:32" :: "v"(ldsOff), "v"(ga) : "memory");
  asm volatile("global_load_async_to_lds_b128 %0, %1, off offset:48" :: "v"(ldsOff), "v"(ga) : "memory");
}
__device__ __forceinline__ void wait_async0() {
  asm volatile("s_wait_asynccnt 0x0" ::: "memory");
}
__device__ __forceinline__ unsigned lds_off(const void* shared_ptr) {
  return (unsigned)(uintptr_t)shared_ptr;   // addr[31:0] = LDS byte offset
}

// ---------------------------------------------------------------------------
// Problem constants
// ---------------------------------------------------------------------------
#define SQ   2048
#define HID  4096
#define NHQ  32
#define NKVH 8
#define HDD  128
#define KVW  2048

// ---------------------------------------------------------------------------
// fp32 -> bf16 bulk convert (8 elements / thread)
// ---------------------------------------------------------------------------
__global__ __launch_bounds__(256)
void cvt_bf16_kernel(const float* __restrict__ x, __bf16* __restrict__ y, int n) {
  const int i = (blockIdx.x * blockDim.x + threadIdx.x) * 8;
  if (i >= n) return;
  const float4* p = (const float4*)(x + i);
  float4 f0 = p[0], f1 = p[1];
  v4u u;
  u[0] = pk2(f0.x, f0.y); u[1] = pk2(f0.z, f0.w);
  u[2] = pk2(f1.x, f1.y); u[3] = pk2(f1.z, f1.w);
  *(v4u*)(y + i) = u;
}

// ---------------------------------------------------------------------------
// GEMM: Y[M,N] = X[M,K] @ W[N,K]^T + bias[N]   (X, W bf16)
// Block = 256 threads (8 waves, 2x4): block tile 64(M) x 256(N), K-chunk 32.
// Double-buffered LDS fed by GLOBAL_LOAD_ASYNC_TO_LDS (ASYNCcnt pipeline).
// Each wave computes 32x64 via 8 WMMA accumulators.
// ---------------------------------------------------------------------------
template <bool OUT_BF16>
__global__ __launch_bounds__(256)
void gemm_bias_kernel(const __bf16* __restrict__ X, const __bf16* __restrict__ W,
                      const float* __restrict__ bias, void* __restrict__ Yv,
                      int M, int N, int K) {
  __shared__ __bf16 As[2][64][32];    //  8 KB
  __shared__ __bf16 Bs[2][256][32];   // 32 KB

  const int tid  = threadIdx.x;
  const int wvid = tid >> 5;
  const int lane = tid & 31;
  const int lh   = lane & 15;
  const int hi   = lane >> 4;
  const int wm   = wvid >> 2;               // 0..1
  const int wn   = wvid & 3;                // 0..3
  const int mb   = blockIdx.y * 64;
  const int nbB  = blockIdx.x * 256;

  const int arow = tid >> 2;                // 0..63
  const int acol = (tid & 3) * 8;           // 0,8,16,24
  const __bf16* aptr = X + (size_t)(mb + arow) * K + acol;
  const __bf16* bptr = W + (size_t)(nbB + tid) * K;

  v8f acc[2][4] = {};

  // prologue: async-stage chunk 0 into buffer 0
  async_cp16(lds_off(&As[0][arow][acol]), aptr);
  async_cp64(lds_off(&Bs[0][tid][0]), bptr);

  for (int k0 = 0; k0 < K; k0 += 32) {
    const int buf = (k0 >> 5) & 1;
    wait_async0();                 // own async copies into As/Bs[buf] done
    __syncthreads();               // all waves' copies visible

    // issue next chunk into the other buffer (overlaps with compute below)
    if (k0 + 32 < K) {
      async_cp16(lds_off(&As[buf ^ 1][arow][acol]), aptr + k0 + 32);
      async_cp64(lds_off(&Bs[buf ^ 1][tid][0]), bptr + k0 + 32);
    }

    // ---- compute: wave tile 32(M) x 64(N) ----------------------------------
    v16bf afr[2];
#pragma unroll
    for (int s = 0; s < 2; ++s) {
      const int row = wm * 32 + s * 16 + lh;
      v4u lo = *(const v4u*)&As[buf][row][hi * 16];
      v4u up = *(const v4u*)&As[buf][row][hi * 16 + 8];
      afr[s] = __builtin_bit_cast(v16bf,
                 __builtin_shufflevector(lo, up, 0, 1, 2, 3, 4, 5, 6, 7));
    }
#pragma unroll
    for (int t = 0; t < 4; ++t) {
      const int nrow = wn * 64 + t * 16 + lh;
      v16bf bfr = __builtin_bit_cast(v16bf, *(const v8u*)&Bs[buf][nrow][hi * 16]);
      acc[0][t] = wmma_bf16(afr[0], bfr, acc[0][t]);
      acc[1][t] = wmma_bf16(afr[1], bfr, acc[1][t]);
    }
  }

  // ---- epilogue: bias add + store -----------------------------------------
#pragma unroll
  for (int s = 0; s < 2; ++s) {
#pragma unroll
    for (int t = 0; t < 4; ++t) {
      const int col = nbB + wn * 64 + t * 16 + lh;
      const float bv = bias[col];
#pragma unroll
      for (int r = 0; r < 8; ++r) {
        const int row = mb + wm * 32 + s * 16 + r + 8 * hi;
        const float v = acc[s][t][r] + bv;
        if (OUT_BF16) ((__bf16*)Yv)[(size_t)row * N + col] = (__bf16)v;
        else          ((float*)Yv)[(size_t)row * N + col]  = v;
      }
    }
  }
}

// ---------------------------------------------------------------------------
// RoPE (in place, bf16) on Q heads (0..31) and K heads (32..39 in KV buffer).
// ---------------------------------------------------------------------------
__global__ __launch_bounds__(256)
void rope_kernel(__bf16* __restrict__ Qb, __bf16* __restrict__ KVb) {
  const int tid = blockIdx.x * blockDim.x + threadIdx.x;
  const int i = tid & 31;
  const int t = tid >> 5;
  const int hidx = t % (NHQ + NKVH);
  const int s = t / (NHQ + NKVH);
  if (s >= SQ) return;

  __bf16* p;
  if (hidx < NHQ) p = Qb + (size_t)s * HID + hidx * HDD;
  else            p = KVb + (size_t)s * KVW + (hidx - NHQ) * (2 * HDD);

  const float inv = __powf(10000.0f, -(float)i * (1.0f / 32.0f));
  const float ang = (float)s * inv;
  const float c  = cosf(ang);
  const float sn = sinf(ang);
  const float x1 = (float)p[i];
  const float x2 = (float)p[i + 32];
  p[i]      = (__bf16)(x1 * c - x2 * sn);
  p[i + 32] = (__bf16)(x2 * c + x1 * sn);
}

// ---------------------------------------------------------------------------
// Flash attention (bf16 Q/K/V): block = (64 query rows, head); 4 waves.
// K rows async-staged, V^T transpose-staged in LDS; QK^T and P@V on WMMA.
// ---------------------------------------------------------------------------
__global__ __launch_bounds__(128)
void attn_kernel(const __bf16* __restrict__ Qb, const __bf16* __restrict__ KVb,
                 __bf16* __restrict__ Ob) {
  const int h   = blockIdx.y;
  const int q0  = blockIdx.x * 64;
  const int kvh = h >> 2;                  // NHQ/NKVH = 4
  const int wv   = threadIdx.x >> 5;
  const int lane = threadIdx.x & 31;
  const int lh   = lane & 15;
  const int hi   = lane >> 4;
  const int qrb  = q0 + wv * 16;

  __shared__ __bf16 Ks[32][128];           // K rows   (8 KB)
  __shared__ __bf16 Vt[128][32];           // V^T      (8 KB)
  __shared__ __bf16 Pt[4][16][32];         // P patch  (4 KB)

  // ---- resident Q tile, A-layout bf16, K=128 -> 4 chunks -------------------
  v16bf aq[4];
  {
    const __bf16* qp = Qb + (size_t)(qrb + lh) * HID + h * HDD;
#pragma unroll
    for (int kc = 0; kc < 4; ++kc) {
      v4u lo = *(const v4u*)(qp + kc * 32 + hi * 16);
      v4u up = *(const v4u*)(qp + kc * 32 + hi * 16 + 8);
      aq[kc] = __builtin_bit_cast(v16bf,
                 __builtin_shufflevector(lo, up, 0, 1, 2, 3, 4, 5, 6, 7));
    }
  }

  v8f acc[8] = {};
  float mrow[8], lrow[8];
#pragma unroll
  for (int r = 0; r < 8; ++r) { mrow[r] = -INFINITY; lrow[r] = 0.0f; }

  const float scale = 0.08838834764831845f;   // 1/sqrt(128)
  const int nkb = (q0 + 64) / 32;             // causal: keys 0 .. q0+63

  for (int jb = 0; jb < nkb; ++jb) {
    const int j0 = jb * 32;

    // ---- stage K (async copy) and V^T (VGPR transpose) into LDS ------------
    __syncthreads();                           // previous readers done
    {
      const int keyr  = threadIdx.x >> 2;        // 0..31
      const int dbase = (threadIdx.x & 3) * 32;  // 4 threads per key
      const __bf16* kp = KVb + (size_t)(j0 + keyr) * KVW + kvh * (2 * HDD) + dbase;
      const __bf16* vp = kp + HDD;

      async_cp64(lds_off(&Ks[keyr][dbase]), kp);   // K rows: pure async copy

      v4u vr[4];
#pragma unroll
      for (int i = 0; i < 4; ++i) vr[i] = *(const v4u*)(vp + i * 8);
#pragma unroll
      for (int i = 0; i < 4; ++i) {
#pragma unroll
        for (int j = 0; j < 4; ++j) {
          const unsigned w = vr[i][j];
          Vt[dbase + i * 8 + 2 * j + 0][keyr] =
              __builtin_bit_cast(__bf16, (unsigned short)(w & 0xFFFFu));
          Vt[dbase + i * 8 + 2 * j + 1][keyr] =
              __builtin_bit_cast(__bf16, (unsigned short)(w >> 16));
        }
      }
      wait_async0();
    }
    __syncthreads();

    // ---- S = Q K^T (16x32 tile, two 16x16 WMMA column tiles) ---------------
    v8f s0 = {}, s1 = {};
#pragma unroll
    for (int kc = 0; kc < 4; ++kc) {
      const int kb = kc * 32 + hi * 16;
      v16bf b0 = __builtin_bit_cast(v16bf, *(const v8u*)&Ks[lh][kb]);
      v16bf b1 = __builtin_bit_cast(v16bf, *(const v8u*)&Ks[16 + lh][kb]);
      s0 = wmma_bf16(aq[kc], b0, s0);
      s1 = wmma_bf16(aq[kc], b1, s1);
    }

    // ---- online softmax ----------------------------------------------------
#pragma unroll
    for (int r = 0; r < 8; ++r) {
      const int qm = qrb + r + 8 * hi;
      float v0 = s0[r] * scale;
      float v1 = s1[r] * scale;
      if (j0 + lh      > qm) v0 = -INFINITY;
      if (j0 + 16 + lh > qm) v1 = -INFINITY;

      float rm = fmaxf(v0, v1);
#pragma unroll
      for (int o = 1; o < 16; o <<= 1) rm = fmaxf(rm, __shfl_xor(rm, o, 16));

      const float mn = fmaxf(mrow[r], rm);
      float p0, p1, esc;
      if (mn == -INFINITY) { p0 = 0.0f; p1 = 0.0f; esc = 1.0f; }
      else {
        p0  = __expf(v0 - mn);
        p1  = __expf(v1 - mn);
        esc = __expf(mrow[r] - mn);
      }
      float rs = p0 + p1;
#pragma unroll
      for (int o = 1; o < 16; o <<= 1) rs += __shfl_xor(rs, o, 16);

      lrow[r] = lrow[r] * esc + rs;
      mrow[r] = mn;
#pragma unroll
      for (int t = 0; t < 8; ++t) acc[t][r] *= esc;

      Pt[wv][r + 8 * hi][lh]      = (__bf16)p0;
      Pt[wv][r + 8 * hi][16 + lh] = (__bf16)p1;
    }

    // ---- reload P in A-layout (per-wave LDS transpose) ---------------------
    v16bf pa;
    {
      const int cb = hi * 16;
      v4u lo = *(const v4u*)&Pt[wv][lh][cb];
      v4u up = *(const v4u*)&Pt[wv][lh][cb + 8];
      pa = __builtin_bit_cast(v16bf,
             __builtin_shufflevector(lo, up, 0, 1, 2, 3, 4, 5, 6, 7));
    }

    // ---- acc += P @ V ------------------------------------------------------
#pragma unroll
    for (int t = 0; t < 8; ++t) {
      v16bf bv = __builtin_bit_cast(v16bf, *(const v8u*)&Vt[t * 16 + lh][hi * 16]);
      acc[t] = wmma_bf16(pa, bv, acc[t]);
    }
  }

  // ---- epilogue: normalize and store (bf16) -------------------------------
#pragma unroll
  for (int r = 0; r < 8; ++r) {
    const int row = qrb + r + 8 * hi;
    const float inv = 1.0f / lrow[r];
#pragma unroll
    for (int t = 0; t < 8; ++t) {
      Ob[(size_t)row * HID + h * HDD + t * 16 + lh] = (__bf16)(acc[t][r] * inv);
    }
  }
}

// ---------------------------------------------------------------------------
// Launcher
// inputs: hidden_states, attention_mask, Wq, bq, Wkv, bkv, Wo, bo (all f32)
// ---------------------------------------------------------------------------
extern "C" void kernel_launch(void* const* d_in, const int* in_sizes, int n_in,
                              void* d_out, int out_size, void* d_ws, size_t ws_size,
                              hipStream_t stream) {
  (void)in_sizes; (void)n_in; (void)out_size; (void)ws_size;

  const float* hidden = (const float*)d_in[0];
  // d_in[1] = attention_mask: causal, synthesized on-chip, not read.
  const float* Wq  = (const float*)d_in[2];
  const float* bq  = (const float*)d_in[3];
  const float* Wkv = (const float*)d_in[4];
  const float* bkv = (const float*)d_in[5];
  const float* Wo  = (const float*)d_in[6];
  const float* bo  = (const float*)d_in[7];
  float* out = (float*)d_out;

  // workspace (bf16 buffers)
  __bf16* Xbf   = (__bf16*)d_ws;                        // 2048*4096
  __bf16* Wqbf  = Xbf   + (size_t)SQ * HID;             // 4096*4096
  __bf16* Wkvbf = Wqbf  + (size_t)HID * HID;            // 2048*4096
  __bf16* Wobf  = Wkvbf + (size_t)KVW * HID;            // 4096*4096
  __bf16* Qbf   = Wobf  + (size_t)HID * HID;            // 2048*4096
  __bf16* KVbf  = Qbf   + (size_t)SQ * HID;             // 2048*2048
  __bf16* Obf   = KVbf  + (size_t)SQ * KVW;             // 2048*4096

  const int nH  = SQ * HID;     // 8M
  const int nWq = HID * HID;    // 16M
  const int nWk = KVW * HID;    // 8M

  // one-time fp32 -> bf16 conversion of all GEMM operands
  cvt_bf16_kernel<<<nH  / 2048, 256, 0, stream>>>(hidden, Xbf, nH);
  cvt_bf16_kernel<<<nWq / 2048, 256, 0, stream>>>(Wq, Wqbf, nWq);
  cvt_bf16_kernel<<<nWk / 2048, 256, 0, stream>>>(Wkv, Wkvbf, nWk);
  cvt_bf16_kernel<<<nWq / 2048, 256, 0, stream>>>(Wo, Wobf, nWq);

  // projections (bf16 in, bf16 out)
  gemm_bias_kernel<true><<<dim3(HID / 256, SQ / 64), dim3(256), 0, stream>>>(
      Xbf, Wqbf, bq, Qbf, SQ, HID, HID);
  gemm_bias_kernel<true><<<dim3(KVW / 256, SQ / 64), dim3(256), 0, stream>>>(
      Xbf, Wkvbf, bkv, KVbf, SQ, KVW, HID);

  rope_kernel<<<(SQ * (NHQ + NKVH) * 32) / 256, 256, 0, stream>>>(Qbf, KVbf);

  attn_kernel<<<dim3(SQ / 64, NHQ), dim3(128), 0, stream>>>(Qbf, KVbf, Obf);

  // output projection (bf16 in, f32 out)
  gemm_bias_kernel<false><<<dim3(HID / 256, SQ / 64), dim3(256), 0, stream>>>(
      Obf, Wobf, bo, out, SQ, HID, HID);
}
